// LoraDecoderLayer_25262997636082
// MI455X (gfx1250) — matrix operand order repaired and checked by reading, not compiled, not run on hardware
//
#include <hip/hip_runtime.h>
#include <math.h>

// ---------------------------------------------------------------------------
// Model dimensions (compile-time constants matching the reference)
// ---------------------------------------------------------------------------
#define BATCH 4
#define SEQ   512          // L == ENC == 512
#define DIM   1024
#define HEADS 16
#define HDIM  64
#define MLPD  4096
#define RANK  32
#define NBUCK 32

typedef __attribute__((ext_vector_type(16))) __bf16 bf16x16;
typedef __attribute__((ext_vector_type(8)))  float  f32x8;

// ===========================================================================
// Elementwise kernels
// ===========================================================================
__global__ void k_cast_bf16(const float* __restrict__ x, __bf16* __restrict__ y,
                            long long n, float scale) {
  long long i = (long long)blockIdx.x * blockDim.x + threadIdx.x;
  if (i < n) y[i] = (__bf16)(x[i] * scale);
}

__global__ void k_add(const float* __restrict__ a, const float* __restrict__ b,
                      float* __restrict__ o, long long n) {
  long long i = (long long)blockIdx.x * blockDim.x + threadIdx.x;
  if (i < n) o[i] = a[i] + b[i];
}

__global__ void k_copy(const float* __restrict__ a, float* __restrict__ o, long long n) {
  long long i = (long long)blockIdx.x * blockDim.x + threadIdx.x;
  if (i < n) o[i] = a[i];
}

__global__ void k_gelu_gate(const float* __restrict__ h0, const float* __restrict__ h1,
                            __bf16* __restrict__ o, long long n) {
  long long i = (long long)blockIdx.x * blockDim.x + threadIdx.x;
  if (i < n) {
    float g = h0[i];
    float v = 0.5f * g * (1.0f + erff(g * 0.70710678118654752f));  // exact gelu
    o[i] = (__bf16)(v * h1[i]);
  }
}

// ===========================================================================
// RMSNorm: one block per row of D=1024, dual fp32 + bf16 output
// ===========================================================================
__global__ void k_rmsnorm(const float* __restrict__ x, const float* __restrict__ g,
                          float* __restrict__ of, __bf16* __restrict__ ob) {
  const int row = blockIdx.x;
  const float* xr = x + (size_t)row * DIM;
  float s = 0.0f;
  for (int i = threadIdx.x; i < DIM; i += blockDim.x) { float v = xr[i]; s += v * v; }
  __shared__ float sh[8];
  for (int o = 16; o; o >>= 1) s += __shfl_xor(s, o, 32);
  if ((threadIdx.x & 31) == 0) sh[threadIdx.x >> 5] = s;
  __syncthreads();
  if (threadIdx.x == 0) {
    float t = 0.0f;
    for (int i = 0; i < 8; ++i) t += sh[i];
    sh[0] = rsqrtf(t * (1.0f / DIM) + 1e-6f);
  }
  __syncthreads();
  const float r = sh[0];
  for (int i = threadIdx.x; i < DIM; i += blockDim.x) {
    float v = xr[i] * r * g[i];
    of[(size_t)row * DIM + i] = v;
    ob[(size_t)row * DIM + i] = (__bf16)v;
  }
}

// ===========================================================================
// Softmax over klen=512 with optional fused T5 relative-position bias + causal
// mask.  blockIdx.x = ((b*H + h)*qlen + q).  256 threads, 2 elems/thread.
// ===========================================================================
__global__ void k_softmax(const float* __restrict__ scores, __bf16* __restrict__ probs,
                          const float* __restrict__ table, int has_bias) {
  const int z = blockIdx.x;
  const int q = z % SEQ;
  const int h = (z / SEQ) % HEADS;
  const float* row = scores + (size_t)z * SEQ;
  __bf16* prow = probs + (size_t)z * SEQ;
  __shared__ float sh[8];

  float v[2];
  float mx = -3.0e38f;
#pragma unroll
  for (int t = 0; t < 2; ++t) {
    int k = threadIdx.x + t * 256;
    float s = row[k];
    if (has_bias) {
      int n = q - k; if (n < 0) n = 0;               // distance into the past
      int bucket;
      if (n < 16) bucket = n;
      else {
        bucket = 16 + (int)(logf((float)n * 0.0625f) * 7.69435163f); // 16/log(8)
        if (bucket > NBUCK - 1) bucket = NBUCK - 1;
      }
      s += table[bucket * HEADS + h];
      if (k > q) s += -1e10f;                         // causal mask
    }
    v[t] = s;
    mx = fmaxf(mx, s);
  }
  for (int o = 16; o; o >>= 1) mx = fmaxf(mx, __shfl_xor(mx, o, 32));
  if ((threadIdx.x & 31) == 0) sh[threadIdx.x >> 5] = mx;
  __syncthreads();
  mx = sh[0];
  for (int i = 1; i < 8; ++i) mx = fmaxf(mx, sh[i]);
  __syncthreads();
  float sum = 0.0f;
#pragma unroll
  for (int t = 0; t < 2; ++t) { v[t] = __expf(v[t] - mx); sum += v[t]; }
  for (int o = 16; o; o >>= 1) sum += __shfl_xor(sum, o, 32);
  if ((threadIdx.x & 31) == 0) sh[threadIdx.x >> 5] = sum;
  __syncthreads();
  sum = 0.0f;
  for (int i = 0; i < 8; ++i) sum += sh[i];
  const float inv = 1.0f / sum;
#pragma unroll
  for (int t = 0; t < 2; ++t) prow[threadIdx.x + t * 256] = (__bf16)(v[t] * inv);
}

// ===========================================================================
// LoRA: low = x @ a      x:[B*Lr, D] fp32,  a:[B,2,D,R] fp32 (sel picks slot)
// ===========================================================================
__global__ void k_lora_low(const float* __restrict__ x, const float* __restrict__ a,
                           float* __restrict__ low, int Lr, int sel) {
  const int row = blockIdx.x;
  const int b = row / Lr;
  const float* xr = x + (size_t)row * DIM;
  const float* ar = a + ((size_t)b * 2 + sel) * DIM * RANK;
  const int r = threadIdx.x;
  float acc = 0.0f;
  for (int d = 0; d < DIM; ++d) acc += xr[d] * ar[d * RANK + r];
  low[(size_t)row * RANK + r] = acc;
}

// C[row, :] += low[row, :] @ b      b:[B,2,R,D]
__global__ void k_lora_acc(float* __restrict__ C, const float* __restrict__ low,
                           const float* __restrict__ bm, int Lr, int sel) {
  const int row = blockIdx.x;
  const int b = row / Lr;
  __shared__ float lw[RANK];
  if (threadIdx.x < RANK) lw[threadIdx.x] = low[(size_t)row * RANK + threadIdx.x];
  __syncthreads();
  const float* br = bm + ((size_t)b * 2 + sel) * RANK * DIM;
  for (int e = threadIdx.x; e < DIM; e += blockDim.x) {
    float s = 0.0f;
#pragma unroll
    for (int r = 0; r < RANK; ++r) s += lw[r] * br[r * DIM + e];
    C[(size_t)row * DIM + e] += s;
  }
}

// ===========================================================================
// Tiled bf16 WMMA GEMM with async Global->LDS double-buffered pipeline.
//   block tile 128x64, 8 waves (4 m x 2 n), wave tile 32x32 = 2x2 16x16x32 WMMA
//   A staged [m][k], B staged [n][k]: fragment reads are b32/b128 DS loads
//   A tile (and B tile when TRANSB) staged with GLOBAL_LOAD_ASYNC_TO_LDS_B128
//   (ASYNCcnt), software-pipelined: issue tile k+1, s_wait_asynccnt for tile k,
//   barrier, compute.  Non-transposed B needs a [k][n]->[n][k] scatter, which
//   async copies can't express, so it keeps the sync load+ds_store path.
// Requirements (all calls satisfy): M%128==0, N%64==0, K%32==0
// ===========================================================================
#define BM 128
#define BN 64
#define BK 32
#define LDAS 40   // padded LDS strides (halfs) to dodge bank conflicts
#define LDBS 40

template <int TRANSB>
__global__ __launch_bounds__(256)
void k_gemm(const __bf16* __restrict__ A, const __bf16* __restrict__ Bm,
            float* __restrict__ C, int K, int lda, int ldb, int ldc,
            int Hb, long long sAb, long long sAh, long long sBb, long long sBh,
            long long sCb, long long sCh, int accum) {
  const int zb = blockIdx.z / Hb, zh = blockIdx.z % Hb;
  A  += zb * sAb + zh * sAh;
  Bm += zb * sBb + zh * sBh;
  C  += zb * sCb + zh * sCh;

  const int m0 = blockIdx.y * BM;
  const int n0 = blockIdx.x * BN;

  __shared__ __bf16 As[2][BM * LDAS];
  __shared__ __bf16 Bs[2][BN * LDBS];

  const int tid  = threadIdx.x;
  const int lane = tid & 31;
  const int wave = tid >> 5;
  const int wm = wave >> 1;          // 0..3  -> 32-row slice
  const int wn = wave & 1;           // 0..1  -> 32-col slice
  const int mh = lane & 15;
  const int hi = lane >> 4;

  f32x8 acc[2][2] = {};

  // A staging: 2 threads per row, 32B (two async b128, offset applied to both
  // the global and the LDS address per the ISA pseudocode)
  const int arow = tid >> 1, aseg = tid & 1;
  const __bf16* Ag = A + (size_t)(m0 + arow) * lda + aseg * 16;
  // B staging (TRANSB): 4 threads per row of B[N,K], 16B each
  const int bn = tid >> 2, kseg = tid & 3;

  // async ops per wave per tile: A=2 (+1 for async B when TRANSB)
  constexpr int NPT = TRANSB ? 3 : 2;

  auto issue_tile = [&](int buf, int k) {
    {
      unsigned la = (unsigned)(size_t)&As[buf][arow * LDAS + aseg * 16];
      unsigned long long ga = (unsigned long long)(size_t)(Ag + k);
      asm volatile("global_load_async_to_lds_b128 %0, %1, off"
                   :: "v"(la), "v"(ga) : "memory");
      asm volatile("global_load_async_to_lds_b128 %0, %1, off offset:16"
                   :: "v"(la), "v"(ga) : "memory");
    }
    if (TRANSB) {
      unsigned lb = (unsigned)(size_t)&Bs[buf][bn * LDBS + kseg * 8];
      unsigned long long gb =
          (unsigned long long)(size_t)(Bm + (size_t)(n0 + bn) * ldb + k + kseg * 8);
      asm volatile("global_load_async_to_lds_b128 %0, %1, off"
                   :: "v"(lb), "v"(gb) : "memory");
    } else {
      const int bk = tid >> 3, nseg = tid & 7;
      uint4 d = *(const uint4*)(Bm + (size_t)(k + bk) * ldb + n0 + nseg * 8);
      const __bf16* hsrc = (const __bf16*)&d;
#pragma unroll
      for (int i = 0; i < 8; ++i) Bs[buf][(nseg * 8 + i) * LDBS + bk] = hsrc[i];
    }
  };

  const int ktiles = K / BK;
  issue_tile(0, 0);                            // prologue: tile 0 -> buffer 0

  for (int kt = 0; kt < ktiles; ++kt) {
    const int cur = kt & 1;
    const bool has_next = (kt + 1) < ktiles;
    if (has_next) {
      __builtin_prefetch(Ag + (kt + 2) * BK, 0, 1);        // warm L2 two ahead
      issue_tile(cur ^ 1, (kt + 1) * BK);                  // pipeline next tile
      asm volatile("s_wait_asynccnt %0" :: "i"(NPT) : "memory");  // tile kt done
    } else {
      asm volatile("s_wait_asynccnt 0" ::: "memory");
    }
    __syncthreads();   // everyone's slice of tile kt is in LDS

    // ---- fragments (ISA 7.12.2 layouts) ----
    union Frag { bf16x16 v; unsigned int u[8]; };
    Frag fa[2], fb[2];
#pragma unroll
    for (int im = 0; im < 2; ++im)
#pragma unroll
      for (int v = 0; v < 8; ++v) {
        const int kk = 16 * (v >> 2) + 8 * hi + 2 * (v & 3);   // A 16x32 layout
        fa[im].u[v] =
            *(const unsigned int*)&As[cur][(wm * 32 + im * 16 + mh) * LDAS + kk];
      }
#pragma unroll
    for (int in_ = 0; in_ < 2; ++in_)
#pragma unroll
      for (int v = 0; v < 8; ++v) {
        const int kk = 16 * (v >> 2) + 8 * hi + 2 * (v & 3);   // B 32x16 layout
        fb[in_].u[v] =
            *(const unsigned int*)&Bs[cur][(wn * 32 + in_ * 16 + mh) * LDBS + kk];
      }
#pragma unroll
    for (int im = 0; im < 2; ++im)
#pragma unroll
      for (int in_ = 0; in_ < 2; ++in_)
        acc[im][in_] = __builtin_amdgcn_wmma_f32_16x16x32_bf16(
            false, fa[im].v, false, fb[in_].v, (short)0, acc[im][in_], false, false);

    __syncthreads();   // done reading buffer `cur`; next iter may overwrite it
  }

  // ---- store (C/D layout: N = lane&15, M = r + 8*hi) ----
#pragma unroll
  for (int im = 0; im < 2; ++im)
#pragma unroll
    for (int in_ = 0; in_ < 2; ++in_) {
      const int ncol = n0 + wn * 32 + in_ * 16 + mh;
#pragma unroll
      for (int r = 0; r < 8; ++r) {
        const int mrow = m0 + wm * 32 + im * 16 + hi * 8 + r;
        float* cp = C + (size_t)mrow * ldc + ncol;
        float val = acc[im][in_][r];
        if (accum) val += *cp;
        *cp = val;
      }
    }
}

// ===========================================================================
// Host side
// ===========================================================================
static void gemm(hipStream_t s, const void* A, const void* B, void* C,
                 int M, int N, int K, int lda, int ldb, int ldc,
                 int batches, int Hb,
                 long long sAb, long long sAh, long long sBb, long long sBh,
                 long long sCb, long long sCh, int transB, int accum) {
  dim3 g(N / BN, M / BM, batches);
  if (transB)
    k_gemm<1><<<g, 256, 0, s>>>((const __bf16*)A, (const __bf16*)B, (float*)C,
                                K, lda, ldb, ldc, Hb, sAb, sAh, sBb, sBh,
                                sCb, sCh, accum);
  else
    k_gemm<0><<<g, 256, 0, s>>>((const __bf16*)A, (const __bf16*)B, (float*)C,
                                K, lda, ldb, ldc, Hb, sAb, sAh, sBb, sBh,
                                sCb, sCh, accum);
}

extern "C" void kernel_launch(void* const* d_in, const int* in_sizes, int n_in,
                              void* d_out, int out_size, void* d_ws, size_t ws_size,
                              hipStream_t stream) {
  (void)in_sizes; (void)n_in; (void)out_size; (void)ws_size;

  const float* inputs  = (const float*)d_in[0];
  const float* encoded = (const float*)d_in[1];
  const float* lqa = (const float*)d_in[2];
  const float* lqb = (const float*)d_in[3];
  const float* lka = (const float*)d_in[4];
  const float* lkb = (const float*)d_in[5];
  const float* lva = (const float*)d_in[6];
  const float* lvb = (const float*)d_in[7];
  const float* loa = (const float*)d_in[8];
  const float* lob = (const float*)d_in[9];
  const float* rel = (const float*)d_in[10];
  const float* ln1 = (const float*)d_in[11];
  const float* ln2 = (const float*)d_in[12];
  const float* ln3 = (const float*)d_in[13];
  const float* sWq = (const float*)d_in[14];
  const float* sWk = (const float*)d_in[15];
  const float* sWv = (const float*)d_in[16];
  const float* sWo = (const float*)d_in[17];
  const float* cWq = (const float*)d_in[18];
  const float* cWk = (const float*)d_in[19];
  const float* cWv = (const float*)d_in[20];
  const float* cWo = (const float*)d_in[21];
  const float* wi0 = (const float*)d_in[22];
  const float* wi1 = (const float*)d_in[23];
  const float* wom = (const float*)d_in[24];

  const long long TOK = (long long)BATCH * SEQ;        // 2048
  const long long ACT = TOK * DIM;                     // 2M elems
  const long long WSZ = (long long)DIM * DIM;          // 1M elems
  const long long MSZ = (long long)DIM * MLPD;         // 4M elems
  const long long SCO = (long long)BATCH * HEADS * SEQ * SEQ;  // 16M elems

  size_t off = 0;
  auto alloc = [&](size_t bytes) -> void* {
    off = (off + 255) & ~(size_t)255;
    void* p = (char*)d_ws + off;
    off += bytes;
    return p;
  };
  __bf16* wq_s = (__bf16*)alloc(WSZ * 2); __bf16* wk_s = (__bf16*)alloc(WSZ * 2);
  __bf16* wv_s = (__bf16*)alloc(WSZ * 2); __bf16* wo_s = (__bf16*)alloc(WSZ * 2);
  __bf16* wq_c = (__bf16*)alloc(WSZ * 2); __bf16* wk_c = (__bf16*)alloc(WSZ * 2);
  __bf16* wv_c = (__bf16*)alloc(WSZ * 2); __bf16* wo_c = (__bf16*)alloc(WSZ * 2);
  __bf16* wi0b = (__bf16*)alloc(MSZ * 2); __bf16* wi1b = (__bf16*)alloc(MSZ * 2);
  __bf16* womb = (__bf16*)alloc(MSZ * 2);
  __bf16* encb = (__bf16*)alloc(ACT * 2);
  float*  xnf  = (float*) alloc(ACT * 4);
  __bf16* xnb  = (__bf16*)alloc(ACT * 2);
  float*  qf   = (float*) alloc(ACT * 4);
  float*  kf   = (float*) alloc(ACT * 4);
  float*  vf   = (float*) alloc(ACT * 4);
  __bf16* qb   = (__bf16*)alloc(ACT * 2);
  __bf16* kb   = (__bf16*)alloc(ACT * 2);
  __bf16* vb   = (__bf16*)alloc(ACT * 2);
  float*  low  = (float*) alloc(TOK * RANK * 4);
  float*  sco  = (float*) alloc(SCO * 4);          // aliased by h0/h1 in MLP
  __bf16* prb  = (__bf16*)alloc(SCO * 2);          // aliased by h_bf in MLP
  float*  ctxf = (float*) alloc(ACT * 4);
  __bf16* ctxb = (__bf16*)alloc(ACT * 2);
  float*  aout = (float*) alloc(ACT * 4);
  float*  xres = (float*) alloc(ACT * 4);
  float*  yres = (float*) alloc(ACT * 4);
  float*  h0   = sco;                              // 8M floats
  float*  h1   = sco + TOK * MLPD;                 // second 8M floats (SCO=16M)
  __bf16* hb   = prb;                              // 8M bf16 inside 16M region

  auto cast = [&](const float* s, __bf16* d, long long n, float sc) {
    k_cast_bf16<<<(unsigned)((n + 255) / 256), 256, 0, stream>>>(s, d, n, sc);
  };

  // ---- weight + encoder casts ----
  cast(sWq, wq_s, WSZ, 1.f); cast(sWk, wk_s, WSZ, 1.f);
  cast(sWv, wv_s, WSZ, 1.f); cast(sWo, wo_s, WSZ, 1.f);
  cast(cWq, wq_c, WSZ, 1.f); cast(cWk, wk_c, WSZ, 1.f);
  cast(cWv, wv_c, WSZ, 1.f); cast(cWo, wo_c, WSZ, 1.f);
  cast(wi0, wi0b, MSZ, 1.f); cast(wi1, wi1b, MSZ, 1.f); cast(wom, womb, MSZ, 1.f);
  cast(encoded, encb, ACT, 1.f);

  const long long sHL = (long long)SEQ * DIM;        // per-batch activation stride
  const long long sBH = (long long)HEADS * SEQ * SEQ;
  const long long sHH = (long long)SEQ * SEQ;

  // ================= self-attention =================
  k_rmsnorm<<<(unsigned)TOK, 256, 0, stream>>>(inputs, ln1, xnf, xnb);
  gemm(stream, xnb, wq_s, qf, TOK, DIM, DIM, DIM, DIM, DIM, 1, 1, 0,0,0,0,0,0, 0, 0);
  k_lora_low<<<(unsigned)TOK, 32, 0, stream>>>(xnf, lqa, low, SEQ, 0);
  k_lora_acc<<<(unsigned)TOK, 256, 0, stream>>>(qf, low, lqb, SEQ, 0);
  gemm(stream, xnb, wk_s, kf, TOK, DIM, DIM, DIM, DIM, DIM, 1, 1, 0,0,0,0,0,0, 0, 0);
  k_lora_low<<<(unsigned)TOK, 32, 0, stream>>>(xnf, lka, low, SEQ, 0);
  k_lora_acc<<<(unsigned)TOK, 256, 0, stream>>>(kf, low, lkb, SEQ, 0);
  gemm(stream, xnb, wv_s, vf, TOK, DIM, DIM, DIM, DIM, DIM, 1, 1, 0,0,0,0,0,0, 0, 0);
  k_lora_low<<<(unsigned)TOK, 32, 0, stream>>>(xnf, lva, low, SEQ, 0);
  k_lora_acc<<<(unsigned)TOK, 256, 0, stream>>>(vf, low, lvb, SEQ, 0);
  cast(qf, qb, ACT, 0.125f);   // 1/sqrt(HD)
  cast(kf, kb, ACT, 1.f);
  cast(vf, vb, ACT, 1.f);
  gemm(stream, qb, kb, sco, SEQ, SEQ, HDIM, DIM, DIM, SEQ,
       BATCH * HEADS, HEADS, sHL, HDIM, sHL, HDIM, sBH, sHH, 1, 0);
  k_softmax<<<(unsigned)(BATCH * HEADS * SEQ), 256, 0, stream>>>(sco, prb, rel, 1);
  gemm(stream, prb, vb, ctxf, SEQ, HDIM, SEQ, SEQ, DIM, DIM,
       BATCH * HEADS, HEADS, sBH, sHH, sHL, HDIM, sHL, HDIM, 0, 0);
  cast(ctxf, ctxb, ACT, 1.f);
  gemm(stream, ctxb, wo_s, aout, TOK, DIM, DIM, DIM, DIM, DIM, 1, 1, 0,0,0,0,0,0, 0, 0);
  k_lora_low<<<(unsigned)TOK, 32, 0, stream>>>(ctxf, loa, low, SEQ, 0);
  k_lora_acc<<<(unsigned)TOK, 256, 0, stream>>>(aout, low, lob, SEQ, 0);
  k_add<<<(unsigned)((ACT + 255) / 256), 256, 0, stream>>>(aout, inputs, xres, ACT);

  // ================= cross-attention =================
  k_rmsnorm<<<(unsigned)TOK, 256, 0, stream>>>(xres, ln2, xnf, xnb);
  gemm(stream, xnb, wq_c, qf, TOK, DIM, DIM, DIM, DIM, DIM, 1, 1, 0,0,0,0,0,0, 0, 0);
  k_lora_low<<<(unsigned)TOK, 32, 0, stream>>>(xnf, lqa, low, SEQ, 1);
  k_lora_acc<<<(unsigned)TOK, 256, 0, stream>>>(qf, low, lqb, SEQ, 1);
  gemm(stream, encb, wk_c, kf, TOK, DIM, DIM, DIM, DIM, DIM, 1, 1, 0,0,0,0,0,0, 0, 0);
  k_lora_low<<<(unsigned)TOK, 32, 0, stream>>>(encoded, lka, low, SEQ, 1);
  k_lora_acc<<<(unsigned)TOK, 256, 0, stream>>>(kf, low, lkb, SEQ, 1);
  gemm(stream, encb, wv_c, vf, TOK, DIM, DIM, DIM, DIM, DIM, 1, 1, 0,0,0,0,0,0, 0, 0);
  k_lora_low<<<(unsigned)TOK, 32, 0, stream>>>(encoded, lva, low, SEQ, 1);
  k_lora_acc<<<(unsigned)TOK, 256, 0, stream>>>(vf, low, lvb, SEQ, 1);
  cast(qf, qb, ACT, 0.125f);
  cast(kf, kb, ACT, 1.f);
  cast(vf, vb, ACT, 1.f);
  gemm(stream, qb, kb, sco, SEQ, SEQ, HDIM, DIM, DIM, SEQ,
       BATCH * HEADS, HEADS, sHL, HDIM, sHL, HDIM, sBH, sHH, 1, 0);
  k_softmax<<<(unsigned)(BATCH * HEADS * SEQ), 256, 0, stream>>>(sco, prb, rel, 0);
  gemm(stream, prb, vb, ctxf, SEQ, HDIM, SEQ, SEQ, DIM, DIM,
       BATCH * HEADS, HEADS, sBH, sHH, sHL, HDIM, sHL, HDIM, 0, 0);
  cast(ctxf, ctxb, ACT, 1.f);
  gemm(stream, ctxb, wo_c, aout, TOK, DIM, DIM, DIM, DIM, DIM, 1, 1, 0,0,0,0,0,0, 0, 0);
  k_lora_low<<<(unsigned)TOK, 32, 0, stream>>>(ctxf, loa, low, SEQ, 1);
  k_lora_acc<<<(unsigned)TOK, 256, 0, stream>>>(aout, low, lob, SEQ, 1);
  k_add<<<(unsigned)((ACT + 255) / 256), 256, 0, stream>>>(aout, xres, yres, ACT);

  // ================= gated-gelu MLP =================
  k_rmsnorm<<<(unsigned)TOK, 256, 0, stream>>>(yres, ln3, xnf, xnb);
  gemm(stream, xnb, wi0b, h0, TOK, MLPD, DIM, DIM, MLPD, MLPD, 1, 1, 0,0,0,0,0,0, 0, 0);
  gemm(stream, xnb, wi1b, h1, TOK, MLPD, DIM, DIM, MLPD, MLPD, 1, 1, 0,0,0,0,0,0, 0, 0);
  const long long HN = TOK * MLPD;
  k_gelu_gate<<<(unsigned)((HN + 255) / 256), 256, 0, stream>>>(h0, h1, hb, HN);
  k_copy<<<(unsigned)((ACT + 255) / 256), 256, 0, stream>>>(yres, (float*)d_out, ACT);
  gemm(stream, hb, womb, d_out, TOK, DIM, MLPD, MLPD, DIM, DIM, 1, 1, 0,0,0,0,0,0, 0, 1);
}
